// MultiAttention_53876069761283
// MI455X (gfx1250) — matrix operand order repaired and checked by reading, not compiled, not run on hardware
//
#include <hip/hip_runtime.h>
#include <hip/hip_bf16.h>
#include <hip/hip_fp16.h>
#include <stdint.h>

typedef __attribute__((ext_vector_type(16))) _Float16 v16h;
typedef __attribute__((ext_vector_type(8)))  _Float16 h8;
typedef __attribute__((ext_vector_type(4)))  _Float16 h4;
typedef __attribute__((ext_vector_type(4)))  float    f4;
typedef __attribute__((ext_vector_type(8)))  float    v8f;
typedef __attribute__((ext_vector_type(4)))  unsigned int u32x4;
typedef __attribute__((ext_vector_type(8)))  int      i32x8;
typedef __attribute__((ext_vector_type(4)))  int      i32x4;

static constexpr int T_  = 1024;
static constexpr int N_  = 8;
static constexpr int D_  = 1024;
static constexpr int H_  = 16;
static constexpr int DK_ = 64;
static constexpr int M_  = T_ * N_;      // 8192 rows for the projections

#define NEG_INF_F (-3.402823466e38f)

// ---------------------------------------------------------------------------
// WMMA fragment helpers — layouts per CDNA5 ISA §7.12.2 (wave32).
// All callers guarantee (p0 + row*ld + kb) is a multiple of 8 halves, so the
// h8 (16-byte) vector loads below are aligned -> ds_load_b128.
// ---------------------------------------------------------------------------

union frag_u {
  v16h v;
  struct { h8 lo; h8 hi; } s;
};

// 16-bit A matrix 16x32 (MxK), LDS row-major with stride `ld` (halves).
__device__ inline v16h load_frag_a(const _Float16* p0, int ld, int lane) {
  const int row = lane & 15;
  const int kb  = (lane >> 4) << 3;   // 0 or 8 (multiple of 8 halves)
  const h8* p = (const h8*)(p0 + row * ld + kb);
  frag_u u;
  u.s.lo = p[0];   // K = kb .. kb+7      -> VGPR0-3
  u.s.hi = p[2];   // K = kb+16 .. kb+23  -> VGPR4-7
  return u.v;
}

// 16-bit B matrix 32x16 (KxN). B kept in LDS as L[n][k] (stride ld), so
// B[k][n] = L[n][k].  lanes 0-15: N=lane, K=0..7,16..23 ; lanes 16-31: K=8..15,24..31.
__device__ inline v16h load_frag_b(const _Float16* p0, int ld, int lane) {
  const int col = lane & 15;
  const int kb  = (lane >> 4) << 3;   // 0 or 8 (multiple of 8 halves)
  const h8* p = (const h8*)(p0 + col * ld + kb);
  frag_u u;
  u.s.lo = p[0];
  u.s.hi = p[2];
  return u.v;
}

__device__ inline v8f wmma16(v16h a, v16h b, v8f c) {
  return __builtin_amdgcn_wmma_f32_16x16x32_f16(false, a, false, b, (short)0, c,
                                                false, false);
}

// ---------------------------------------------------------------------------
// Tensor Data Mover: DMA a 64x64 f16 tile (row stride `g_stride` halves in
// global) into LDS with an 8-half pad after each 64-half row (stride 72).
// D# layout per CDNA5 ISA. Issued by one wave.
// This toolchain exposes the 6-arg builtin (extra int32x8 group).
// ---------------------------------------------------------------------------
__device__ inline void tdm_load_64x64_f16(const _Float16* gsrc, uint32_t lds_off,
                                          uint32_t g_stride_elems) {
  const uint64_t ga = (uint64_t)(uintptr_t)gsrc;
  u32x4 g0;
  g0.x = 1u;                                        // count=1, user desc
  g0.y = lds_off;                                   // lds_addr (bytes)
  g0.z = (uint32_t)(ga & 0xffffffffu);              // global_addr[95:64]
  g0.w = (uint32_t)((ga >> 32) & 0x01ffffffu)       // global_addr[120:96]
         | (2u << 30);                              // type = 2 ("image")
  i32x8 g1;
  g1[0] = (1 << 16)        // data_size = 1 -> 2 bytes
        | (1 << 20)        // pad_enable
        | (4 << 22)        // pad_interval: 32 DWORDs (one 64-half row)
        | (3 << 25);       // pad_amount:   4 DWORDs (8 halves)
  g1[1] = (int)(64u << 16);        // tensor_dim0[15:0] = 64
  g1[2] = (int)(64u << 16);        // tensor_dim0 hi = 0; tensor_dim1 lo = 64
  g1[3] = (int)(64u << 16);        // tensor_dim1 hi = 0; tile_dim0 = 64
  g1[4] = 64;                      // tile_dim1 = 64; tile_dim2 = 0
  g1[5] = (int)g_stride_elems;     // tensor_dim0_stride[31:0]
  g1[6] = 0;                       // stride hi; tensor_dim1_stride lo = 0
  g1[7] = 0;
  const i32x4 z4 = {0, 0, 0, 0};             // groups 2/3 unused (2-D tensor)
  const i32x8 z8 = {0, 0, 0, 0, 0, 0, 0, 0};
  __builtin_amdgcn_tensor_load_to_lds(g0, g1, z4, z4, z8, 0);
}

// ---------------------------------------------------------------------------
// GEMM with bias:  out[m][n] = sum_k A[m][k] * W[n][k] + bias[n]
// ---------------------------------------------------------------------------
template <bool A_HALF, bool OUT_HALF>
__global__ __launch_bounds__(256)
void gemm_bias_kernel(const void* __restrict__ Aptr,
                      const float* __restrict__ W,
                      const float* __restrict__ bias,
                      void* __restrict__ outp, int M, int N, int K) {
  constexpr int BM = 128, BN = 64, BK = 32;
  constexpr int LDA = BK + 8, LDB = BK + 8;   // 40 halves (mult. of 8)
  __shared__ _Float16 sA[BM * LDA];
  __shared__ _Float16 sB[BN * LDB];

  const int tid  = threadIdx.x;
  const int lane = tid & 31;
  const int w    = tid >> 5;
  const int wm   = w & 3;
  const int wn   = w >> 2;
  const int m0   = blockIdx.y * BM;
  const int n0   = blockIdx.x * BN;

  v8f c[2][2] = {{{}, {}}, {{}, {}}};

  for (int k0 = 0; k0 < K; k0 += BK) {
    if (A_HALF) {
      const _Float16* A = (const _Float16*)Aptr;
#pragma unroll
      for (int i = 0; i < 2; ++i) {
        int idx = tid + 256 * i;
        int r = idx >> 2;
        int g = (idx & 3) * 8;
        h8 v = *(const h8*)(A + (size_t)(m0 + r) * K + k0 + g);
        *(h8*)(&sA[r * LDA + g]) = v;
      }
    } else {
      const float* A = (const float*)Aptr;
#pragma unroll
      for (int i = 0; i < 4; ++i) {
        int idx = tid + 256 * i;
        int r = idx >> 3;
        int g = (idx & 7) * 4;
        f4 v = *(const f4*)(A + (size_t)(m0 + r) * K + k0 + g);
        *(h4*)(&sA[r * LDA + g]) = __builtin_convertvector(v, h4);
      }
    }
#pragma unroll
    for (int i = 0; i < 2; ++i) {
      int idx = tid + 256 * i;
      int r = idx >> 3;
      int g = (idx & 7) * 4;
      f4 v = *(const f4*)(W + (size_t)(n0 + r) * K + k0 + g);
      *(h4*)(&sB[r * LDB + g]) = __builtin_convertvector(v, h4);
    }
    __syncthreads();

    v16h a0 = load_frag_a(&sA[(wm * 32 + 0)  * LDA], LDA, lane);
    v16h a1 = load_frag_a(&sA[(wm * 32 + 16) * LDA], LDA, lane);
    v16h b0 = load_frag_b(&sB[(wn * 32 + 0)  * LDB], LDB, lane);
    v16h b1 = load_frag_b(&sB[(wn * 32 + 16) * LDB], LDB, lane);
    c[0][0] = wmma16(a0, b0, c[0][0]);
    c[0][1] = wmma16(a0, b1, c[0][1]);
    c[1][0] = wmma16(a1, b0, c[1][0]);
    c[1][1] = wmma16(a1, b1, c[1][1]);
    __syncthreads();
  }

  const int hi  = lane >> 4;
  const int col = lane & 15;
#pragma unroll
  for (int mi = 0; mi < 2; ++mi) {
#pragma unroll
    for (int ni = 0; ni < 2; ++ni) {
      int gn = n0 + wn * 32 + ni * 16 + col;
      float bv = bias[gn];
#pragma unroll
      for (int v = 0; v < 8; ++v) {
        int gm = m0 + wm * 32 + mi * 16 + hi * 8 + v;
        float val = c[mi][ni][v] + bv;
        if (OUT_HALF)
          ((_Float16*)outp)[(size_t)gm * N + gn] = (_Float16)val;
        else
          ((float*)outp)[(size_t)gm * N + gn] = val;
      }
    }
  }
}

// ---------------------------------------------------------------------------
// Flash-style attention. Block = (q-tile of 64, batch n, head h), 256 threads.
// Q and K tiles staged to LDS via the Tensor Data Mover; V staged transposed
// by vector lanes (TDM cannot transpose).
// ---------------------------------------------------------------------------
__global__ __launch_bounds__(256)
void attn_kernel(const _Float16* __restrict__ qkv,
                 const unsigned char* __restrict__ kpad,   // [N][T] bool
                 _Float16* __restrict__ attn_out) {        // [T][N][D] f16
  constexpr int LDH  = DK_ + 8;  // 72 halves (mult. of 8)
  constexpr int LDSS = 64 + 8;   // f32 stride
  __shared__ _Float16 sQ [64 * LDH];
  __shared__ _Float16 sK [64 * LDH];
  __shared__ _Float16 sVt[64 * LDH];   // transposed: sVt[dk][key]
  __shared__ _Float16 sP [64 * LDH];
  __shared__ float    sS [64 * LDSS];
  __shared__ float    sScale[64];
  __shared__ float    sL[64];

  const int tid  = threadIdx.x;
  const int lane = tid & 31;
  const int w    = tid >> 5;
  const int wm   = w & 3;      // wave-row: 16 q-rows each
  const int wn   = w >> 2;     // wave-col: 32 cols each
  const int qb   = blockIdx.x;
  const int n    = blockIdx.y;
  const int h    = blockIdx.z;
  const int q0   = qb * 64;
  const float scale = 0.125f;  // 1/sqrt(64)
  const uint32_t row_stride = (uint32_t)(N_ * 3 * D_);  // halves between rows

  const int hi = lane >> 4;
  const int cl = lane & 15;

  // ---- Q tile (64x64 f16) via TDM, issued by wave 0 ----
  if (w == 0) {
    tdm_load_64x64_f16(qkv + ((size_t)q0 * N_ + n) * (3 * D_) + h * DK_,
                       (uint32_t)(uintptr_t)(void*)sQ, row_stride);
  }

  float m_r = NEG_INF_F, l_r = 0.0f;   // live in threads tid<64 (row = tid)
  v8f o0 = {}, o1 = {};

  for (int it = 0; it <= qb; ++it) {
    const int s0 = it * 64;
    __syncthreads();   // prior iteration done reading sK/sVt/sS/sP

    // ---- K tile via TDM (wave 0) ----
    if (w == 0) {
      tdm_load_64x64_f16(qkv + ((size_t)s0 * N_ + n) * (3 * D_) + D_ + h * DK_,
                         (uint32_t)(uintptr_t)(void*)sK, row_stride);
    }

    // ---- V tile, transposed, by all lanes ----
#pragma unroll
    for (int i = 0; i < 2; ++i) {
      int idx = tid + 256 * i;
      int r = idx >> 3;
      int g = (idx & 7) * 8;
      h8 vv = *(const h8*)(qkv + ((size_t)(s0 + r) * N_ + n) * (3 * D_)
                               + 2 * D_ + h * DK_ + g);
#pragma unroll
      for (int j = 0; j < 8; ++j) sVt[(g + j) * LDH + r] = vv[j];
    }
    if (it < qb) {  // hint next V tile toward the caches (global_prefetch_b8)
      __builtin_prefetch(qkv + ((size_t)(s0 + 64 + (tid >> 2)) * N_ + n) * (3 * D_)
                             + 2 * D_ + h * DK_, 0, 0);
    }
    if (w == 0) __builtin_amdgcn_s_wait_tensorcnt(0);  // Q (first iter) + K done
    __syncthreads();

    // ---- S = scale * Q K^T ----
    v8f c0 = {}, c1 = {};
#pragma unroll
    for (int kk = 0; kk < DK_; kk += 32) {
      v16h a  = load_frag_a(&sQ[(wm * 16) * LDH + kk], LDH, lane);
      v16h b0 = load_frag_b(&sK[(wn * 32 + 0)  * LDH + kk], LDH, lane);
      v16h b1 = load_frag_b(&sK[(wn * 32 + 16) * LDH + kk], LDH, lane);
      c0 = wmma16(a, b0, c0);
      c1 = wmma16(a, b1, c1);
    }

    // ---- masked write of S into LDS ----
    const int colL0 = wn * 32 + cl, colL1 = colL0 + 16;
    const int colG0 = s0 + colL0,   colG1 = s0 + colL1;
    const unsigned char kp0 = kpad[(size_t)n * T_ + colG0];
    const unsigned char kp1 = kpad[(size_t)n * T_ + colG1];
#pragma unroll
    for (int v = 0; v < 8; ++v) {
      int rL = wm * 16 + hi * 8 + v;
      int rG = q0 + rL;
      float v0 = c0[v] * scale;
      float v1 = c1[v] * scale;
      if (colG0 > rG || kp0) v0 = NEG_INF_F;   // causal + key padding
      if (colG1 > rG || kp1) v1 = NEG_INF_F;
      sS[rL * LDSS + colL0] = v0;
      sS[rL * LDSS + colL1] = v1;
    }
    __syncthreads();

    // ---- online softmax: one scanner thread per q-row ----
    if (tid < 64) {
      const int r = tid;
      float mt = NEG_INF_F;
#pragma unroll 8
      for (int j = 0; j < 64; ++j) mt = fmaxf(mt, sS[r * LDSS + j]);
      float mnew = fmaxf(m_r, mt);
      float corr = __expf(m_r - mnew);
      float ssum = 0.0f;
#pragma unroll 8
      for (int j = 0; j < 64; ++j) {
        float p = __expf(sS[r * LDSS + j] - mnew);
        sP[r * LDH + j] = (_Float16)p;
        ssum += p;
      }
      l_r = l_r * corr + ssum;
      m_r = mnew;
      sScale[r] = corr;
    }
    __syncthreads();

    // ---- rescale O fragments ----
#pragma unroll
    for (int v = 0; v < 8; ++v) {
      float cr = sScale[wm * 16 + hi * 8 + v];
      o0[v] *= cr;
      o1[v] *= cr;
    }

    // ---- O += P @ V ----
#pragma unroll
    for (int kk = 0; kk < 64; kk += 32) {
      v16h a  = load_frag_a(&sP[(wm * 16) * LDH + kk], LDH, lane);
      v16h b0 = load_frag_b(&sVt[(wn * 32 + 0)  * LDH + kk], LDH, lane);
      v16h b1 = load_frag_b(&sVt[(wn * 32 + 16) * LDH + kk], LDH, lane);
      o0 = wmma16(a, b0, o0);
      o1 = wmma16(a, b1, o1);
    }
  }

  // ---- finalize: divide by l, write f16 attn output ----
  if (tid < 64) sL[tid] = l_r;
  __syncthreads();
#pragma unroll
  for (int v = 0; v < 8; ++v) {
    int rL = wm * 16 + hi * 8 + v;
    int rG = q0 + rL;
    float inv = 1.0f / sL[rL];
    size_t base = ((size_t)rG * N_ + n) * D_ + h * DK_;
    attn_out[base + wn * 32 + cl]      = (_Float16)(o0[v] * inv);
    attn_out[base + wn * 32 + 16 + cl] = (_Float16)(o1[v] * inv);
  }
}

// ---------------------------------------------------------------------------
// Launcher
// ---------------------------------------------------------------------------
extern "C" void kernel_launch(void* const* d_in, const int* in_sizes, int n_in,
                              void* d_out, int out_size, void* d_ws, size_t ws_size,
                              hipStream_t stream) {
  const float* x    = (const float*)d_in[0];   // [T][N][D]
  const float* w_in = (const float*)d_in[1];   // [3D][D]
  const float* b_in = (const float*)d_in[2];   // [3D]
  const float* w_o  = (const float*)d_in[3];   // [D][D]
  const float* b_o  = (const float*)d_in[4];   // [D]
  // d_in[5] = attn_mask (causal) — recomputed analytically in-kernel
  const unsigned char* kpad = (const unsigned char*)d_in[6];  // [N][T] bool
  float* out = (float*)d_out;                  // [T][N][D] f32

  _Float16* qkv_ws  = (_Float16*)d_ws;                       // [T*N][3D] f16
  _Float16* attn_ws = qkv_ws + (size_t)M_ * 3 * D_;          // [T*N][D]  f16
  if (ws_size < ((size_t)M_ * 3 * D_ + (size_t)M_ * D_) * sizeof(_Float16))
    return;

  gemm_bias_kernel<false, true>
      <<<dim3((3 * D_) / 64, M_ / 128), 256, 0, stream>>>(
          x, w_in, b_in, qkv_ws, M_, 3 * D_, D_);

  attn_kernel<<<dim3(T_ / 64, N_, H_), 256, 0, stream>>>(qkv_ws, kpad, attn_ws);

  gemm_bias_kernel<true, false>
      <<<dim3(D_ / 64, M_ / 128), 256, 0, stream>>>(
          attn_ws, w_o, b_o, out, M_, D_, D_);
}
